// EncoderBlock_44968307589614
// MI455X (gfx1250) — matrix operand order repaired and checked by reading, not compiled, not run on hardware
//
#include <hip/hip_runtime.h>
#include <hip/hip_bf16.h>
#include <math.h>

// ---------------- dims ----------------
#define Bq   8
#define Cc   64
#define Tt   512
#define Ff   64
#define Hh   4
#define Dd   1024      // E*F
#define TF   (Tt*Ff)   // 32768
#define RC   256       // R*C
#define NLAY 6
#define CO   128
#define FOUT 32

typedef __attribute__((ext_vector_type(16))) __bf16 v16bf;
typedef __attribute__((ext_vector_type(8)))  __bf16 v8bf;
typedef __attribute__((ext_vector_type(8)))  float  v8f;

// ---------------- CDNA5 helpers ----------------
__device__ __forceinline__ v8f wmma_bf16(v16bf a, v16bf b, v8f c) {
#if defined(__gfx1250__)
  return __builtin_amdgcn_wmma_f32_16x16x32_bf16(false, a, false, b, (short)0, c, false, false);
#else
  return c;
#endif
}

__device__ __forceinline__ float fast_rcp(float x) {
#if defined(__gfx1250__)
  return __builtin_amdgcn_rcpf(x);
#else
  return 1.0f / x;
#endif
}

// async global->LDS copy (16B), tracked by ASYNCcnt
__device__ __forceinline__ void async_ld_b128(unsigned lds_off, const void* gptr) {
#if defined(__gfx1250__)
  asm volatile("global_load_async_to_lds_b128 %0, %1, off"
               :: "v"(lds_off), "v"((unsigned long long)(uintptr_t)gptr)
               : "memory");
#else
  (void)lds_off; (void)gptr;
#endif
}
__device__ __forceinline__ void wait_async0() {
#if defined(__gfx1250__)
  asm volatile("s_wait_asynccnt 0" ::: "memory");
#endif
}

// branch-free exact-ish GELU: 0.5*h*(1+erf(h/sqrt(2))), A&S 7.1.26 erf (|err|<1.5e-7)
__device__ __forceinline__ float gelu_f(float h) {
  float xs = h * 0.70710678118f;
  float ax = fabsf(xs);
  float t  = fast_rcp(1.0f + 0.3275911f * ax);
  float p  = t * (0.254829592f + t * (-0.284496736f +
             t * (1.421413741f + t * (-1.453152027f + t * 1.061405429f))));
  float r  = 1.0f - p * __expf(-ax * ax);
  float er = copysignf(r, xs);
  return 0.5f * h * (1.0f + er);
}

// A-matrix fragment (16x32 bf16), row-major [M][K], row stride rs.
// lane<16: K {0..7,16..23}; lane>=16: K {8..15,24..31}
__device__ __forceinline__ v16bf frag_a(const __bf16* base, int row, int rs, int k0, int lane) {
  const __bf16* p = base + (long)row * rs + k0 + ((lane & 16) ? 8 : 0);
  union { v16bf v; v8bf h[2]; } u;
  u.h[0] = *(const v8bf*)(p);
  u.h[1] = *(const v8bf*)(p + 16);
  return u.v;
}
// B-matrix fragment (32x16 bf16), row-major [N][K], row stride rs.
// lane<16: K {0..7,16..23}; lane>=16: K {8..15,24..31}
__device__ __forceinline__ v16bf frag_b(const __bf16* base, int col, int rs, int k0, int lane) {
  const __bf16* p = base + (long)col * rs + k0 + ((lane & 16) ? 8 : 0);
  union { v16bf v; v8bf h[2]; } u;
  u.h[0] = *(const v8bf*)(p);
  u.h[1] = *(const v8bf*)(p + 16);
  return u.v;
}

// ---------------- weight fp32 -> bf16 ----------------
__global__ void k_cvt(const float* __restrict__ s, __bf16* __restrict__ d, int n) {
  int i = blockIdx.x * 256 + threadIdx.x;
  if (i < n) d[i] = (__bf16)s[i];
}

// ---------------- TFCM depthwise dilated conv ----------------
__global__ __launch_bounds__(64) void k_tfcm_dw(
    const float* __restrict__ x, float* __restrict__ y,
    const float* __restrict__ w, const float* __restrict__ bias, int dil)
{
  int f = threadIdx.x;
  int t = blockIdx.x & (Tt - 1);
  int c = (blockIdx.x >> 9) & (Cc - 1);
  int b = blockIdx.x >> 15;
  const float* xb = x + ((long)b * Cc + c) * (long)TF;
  float acc = bias[c];
  #pragma unroll
  for (int kt = 0; kt < 2; kt++) {
    int ti = t + kt * dil - dil;
    if (ti < 0) continue;
    const float* xr = xb + (long)ti * Ff;
    #pragma unroll
    for (int kf = 0; kf < 7; kf++) {
      int fi = f + kf - 3;
      float xv = (fi >= 0 && fi < Ff) ? xr[fi] : 0.0f;
      acc += w[(c * 2 + kt) * 7 + kf] * xv;
    }
  }
  y[(((long)b * Cc + c) * Tt + t) * Ff + f] = acc;
}

// ---------------- TFCM: LN + pw1 + GELU + pw2 + residual ----------------
__global__ __launch_bounds__(256) void k_tfcm_mlp(
    const float* __restrict__ y, float* __restrict__ x,
    const float* __restrict__ lnw, const float* __restrict__ lnb,
    const __bf16* __restrict__ w1, const float* __restrict__ b1,
    const __bf16* __restrict__ w2, const float* __restrict__ b2)
{
  __shared__ float  ls_y[64][68];   // [c][pos], rows 272B (16B aligned) for async b128
  __shared__ __bf16 ls_a[64][Cc];   // [pos][c] normalized
  __shared__ __bf16 ls_h[64][RC];   // [pos][n] hidden
  int tid = threadIdx.x, lane = tid & 31, wave = tid >> 5;
  long p0 = (long)blockIdx.x * 64;
  int b  = (int)(p0 / TF);
  int s0 = (int)(p0 % TF);
  const float* ybase = y + (long)b * Cc * TF + s0;

  // async stage: 64 channels x 16 chunks of 4 floats
  for (int idx = tid; idx < 64 * 16; idx += 256) {
    int c = idx >> 4, ch = idx & 15;
    async_ld_b128((unsigned)(uintptr_t)(&ls_y[c][ch * 4]),
                  ybase + (long)c * TF + ch * 4);
  }
  wait_async0();
  __syncthreads();

  // LayerNorm over channels per position
  if (tid < 64) {
    int pos = tid;
    float mu = 0.f;
    for (int c = 0; c < Cc; c++) mu += ls_y[c][pos];
    mu *= (1.0f / Cc);
    float var = 0.f;
    for (int c = 0; c < Cc; c++) { float d = ls_y[c][pos] - mu; var += d * d; }
    var *= (1.0f / Cc);
    float rs = rsqrtf(var + 1e-6f);
    for (int c = 0; c < Cc; c++)
      ls_a[pos][c] = (__bf16)((ls_y[c][pos] - mu) * rs * lnw[c] + lnb[c]);
  }
  __syncthreads();

  // pw1: M=64 N=256 K=64. wave: mt = wave&3, n-tiles (wave>>2)*8 + j
  {
    int mt = wave & 3, nth = wave >> 2;
    v8f acc[8];
    #pragma unroll
    for (int j = 0; j < 8; j++) {
      float bi = b1[(nth * 8 + j) * 16 + (lane & 15)];
      #pragma unroll
      for (int r = 0; r < 8; r++) acc[j][r] = bi;
    }
    #pragma unroll
    for (int ks = 0; ks < 2; ks++) {
      v16bf a = frag_a(&ls_a[0][0], mt * 16 + (lane & 15), Cc, ks * 32, lane);
      #pragma unroll
      for (int j = 0; j < 8; j++) {
        v16bf bm = frag_b(w1, (nth * 8 + j) * 16 + (lane & 15), Cc, ks * 32, lane);
        acc[j] = wmma_bf16(a, bm, acc[j]);
      }
    }
    int rbase = mt * 16 + ((lane & 16) ? 8 : 0);
    #pragma unroll
    for (int j = 0; j < 8; j++) {
      int n = (nth * 8 + j) * 16 + (lane & 15);
      #pragma unroll
      for (int r = 0; r < 8; r++)
        ls_h[rbase + r][n] = (__bf16)gelu_f(acc[j][r]);
    }
  }
  __syncthreads();

  // pw2: M=64 N=64 K=256. wave: mt = wave&3, n-tiles (wave>>2)*2 + j
  {
    int mt = wave & 3, nb = wave >> 2;
    float* xbase = x + (long)b * Cc * TF + s0;
    v8f acc[2];
    #pragma unroll
    for (int j = 0; j < 2; j++) {
      float bi = b2[(nb * 2 + j) * 16 + (lane & 15)];
      #pragma unroll
      for (int r = 0; r < 8; r++) acc[j][r] = bi;
    }
    #pragma unroll
    for (int ks = 0; ks < 8; ks++) {
      v16bf a = frag_a(&ls_h[0][0], mt * 16 + (lane & 15), RC, ks * 32, lane);
      #pragma unroll
      for (int j = 0; j < 2; j++) {
        v16bf bm = frag_b(w2, (nb * 2 + j) * 16 + (lane & 15), RC, ks * 32, lane);
        acc[j] = wmma_bf16(a, bm, acc[j]);
      }
    }
    int rbase = mt * 16 + ((lane & 16) ? 8 : 0);
    #pragma unroll
    for (int j = 0; j < 2; j++) {
      int n = (nb * 2 + j) * 16 + (lane & 15);
      #pragma unroll
      for (int r = 0; r < 8; r++) {
        long off = (long)n * TF + (rbase + r);
        xbase[off] += acc[j][r];
      }
    }
  }
}

// ---------------- q/k/v projection: relu(W@x) + LN4DCF + head reshape ----------------
__global__ __launch_bounds__(256) void k_proj(
    const float* __restrict__ x, const __bf16* __restrict__ w,
    const float* __restrict__ g, const float* __restrict__ bb,
    __bf16* __restrict__ out, int transposed)
{
  __shared__ __bf16 ls_a[64][Cc];     // [f][c]
  __shared__ float  ls_t[64][Cc + 1]; // [f][o]
  __shared__ float  red[256], red2[256];
  int tid = threadIdx.x, lane = tid & 31, wave = tid >> 5;
  int t = blockIdx.x & (Tt - 1);
  int b = blockIdx.x >> 9;
  const float* xb = x + (long)b * Cc * TF + (long)t * Ff;

  for (int idx = tid; idx < 64 * Cc; idx += 256) {
    int c = idx >> 6, f = idx & 63;
    ls_a[f][c] = (__bf16)xb[(long)c * TF + f];
  }
  __syncthreads();

  // GEMM M=64(f) N=64(o) K=64(c): wave: mt = wave&3, n-tiles (wave>>2)*2 + j
  {
    int mt = wave & 3, nb = wave >> 2;
    v8f acc[2];
    #pragma unroll
    for (int j = 0; j < 2; j++)
      #pragma unroll
      for (int r = 0; r < 8; r++) acc[j][r] = 0.f;
    #pragma unroll
    for (int ks = 0; ks < 2; ks++) {
      v16bf a = frag_a(&ls_a[0][0], mt * 16 + (lane & 15), Cc, ks * 32, lane);
      #pragma unroll
      for (int j = 0; j < 2; j++) {
        v16bf bm = frag_b(w, (nb * 2 + j) * 16 + (lane & 15), Cc, ks * 32, lane);
        acc[j] = wmma_bf16(a, bm, acc[j]);
      }
    }
    int rbase = mt * 16 + ((lane & 16) ? 8 : 0);
    #pragma unroll
    for (int j = 0; j < 2; j++) {
      int n = (nb * 2 + j) * 16 + (lane & 15);
      #pragma unroll
      for (int r = 0; r < 8; r++)
        ls_t[rbase + r][n] = fmaxf(acc[j][r], 0.0f);
    }
  }
  __syncthreads();

  // LN4DCF over the whole 64x64 (C,F) tile
  float ps = 0.f, ps2 = 0.f;
  for (int j = 0; j < 16; j++) {
    int idx = tid * 16 + j;
    float v = ls_t[idx >> 6][idx & 63];
    ps += v; ps2 += v * v;
  }
  red[tid] = ps; red2[tid] = ps2;
  __syncthreads();
  for (int st = 128; st > 0; st >>= 1) {
    if (tid < st) { red[tid] += red[tid + st]; red2[tid] += red2[tid + st]; }
    __syncthreads();
  }
  float mu  = red[0] * (1.0f / 4096.0f);
  float var = red2[0] * (1.0f / 4096.0f) - mu * mu;
  float inv = fast_rcp(sqrtf(var + 1e-5f) + 1e-5f);

  for (int idx = tid; idx < 4096; idx += 256) {
    int f = idx >> 6, o = idx & 63;
    float v = (ls_t[f][o] - mu) * inv * g[o * Ff + f] + bb[o * Ff + f];
    int h = o >> 4, e = o & 15;
    int d = e * Ff + f;
    long off = transposed
             ? ((((long)b * Hh + h) * Dd + d) * Tt + t)
             : ((((long)b * Hh + h) * Tt + t) * Dd + d);
    out[off] = (__bf16)v;
  }
}

// ---------------- causal attention: S=QK^T/32, softmax, O=PV ----------------
__global__ __launch_bounds__(256) void k_attn(
    const __bf16* __restrict__ q, const __bf16* __restrict__ k,
    const __bf16* __restrict__ vt, float* __restrict__ s_out)
{
  __shared__ float  ls_s[16][Tt];
  __shared__ __bf16 ls_p[16][Tt];
  __shared__ float  pm[16][16];
  __shared__ float  ls_m[16], ls_sum[16];
  int tid = threadIdx.x, lane = tid & 31, wave = tid >> 5;
  int mtq = blockIdx.x & 31;
  int bh  = blockIdx.x >> 5;
  int tq0 = mtq * 16;
  const __bf16* qb = q  + (long)bh * Tt * Dd;
  const __bf16* kb = k  + (long)bh * Tt * Dd;
  const __bf16* vb = vt + (long)bh * Dd * Tt;

  // S = Q K^T: wave owns 4 col-tiles nt = wave + 8j; skip fully-masked (nt > mtq)
  {
    v8f acc[4];
    #pragma unroll
    for (int j = 0; j < 4; j++)
      #pragma unroll
      for (int r = 0; r < 8; r++) acc[j][r] = 0.f;
    const __bf16* arow = qb + (long)(tq0 + (lane & 15)) * Dd;
    for (int ks = 0; ks < 32; ks++) {
      if (ks + 1 < 32) __builtin_prefetch(arow + (ks + 1) * 32, 0, 0);
      v16bf a = frag_a(qb, tq0 + (lane & 15), Dd, ks * 32, lane);
      #pragma unroll
      for (int j = 0; j < 4; j++) {
        int nt = wave + 8 * j;
        if (nt <= mtq) {                       // wave-uniform causal skip
          v16bf bm = frag_b(kb, nt * 16 + (lane & 15), Dd, ks * 32, lane);
          acc[j] = wmma_bf16(a, bm, acc[j]);
        }
      }
    }
    int rbase = (lane & 16) ? 8 : 0;
    #pragma unroll
    for (int j = 0; j < 4; j++) {
      int nt = wave + 8 * j;
      int col = nt * 16 + (lane & 15);
      #pragma unroll
      for (int r = 0; r < 8; r++) {
        int row = rbase + r;
        float sv = -1e9f;
        if (nt <= mtq) {
          sv = acc[j][r] * (1.0f / 32.0f);
          if (col > tq0 + row) sv = -1e9f;
        }
        ls_s[row][col] = sv;
      }
    }
  }
  __syncthreads();

  // softmax per row (16 threads/row, 32 cols each)
  {
    int row = tid >> 4, seg = tid & 15;
    float m = -1e30f;
    for (int c2 = seg * 32; c2 < seg * 32 + 32; c2++) m = fmaxf(m, ls_s[row][c2]);
    pm[row][seg] = m;
    __syncthreads();
    if (seg == 0) { float mm = -1e30f; for (int j2 = 0; j2 < 16; j2++) mm = fmaxf(mm, pm[row][j2]); ls_m[row] = mm; }
    __syncthreads();
    float mrow = ls_m[row], ssum = 0.f;
    for (int c2 = seg * 32; c2 < seg * 32 + 32; c2++) {
      float e2 = __expf(ls_s[row][c2] - mrow);
      ls_s[row][c2] = e2; ssum += e2;
    }
    pm[row][seg] = ssum;
    __syncthreads();
    if (seg == 0) { float st = 0.f; for (int j2 = 0; j2 < 16; j2++) st += pm[row][j2]; ls_sum[row] = st; }
    __syncthreads();
    float inv = fast_rcp(ls_sum[row]);
    for (int c2 = seg * 32; c2 < seg * 32 + 32; c2++)
      ls_p[row][c2] = (__bf16)(ls_s[row][c2] * inv);
  }
  __syncthreads();

  // O = P @ V: M=16 N=1024 K<=tq0+16 (rest of P is zero); wave owns 8 d-tiles
  {
    int b = bh >> 2, h = bh & 3;
    int ksmax = (tq0 + 16 + 31) >> 5;          // causal clip of K loop
    v8f acc[8];
    #pragma unroll
    for (int j = 0; j < 8; j++)
      #pragma unroll
      for (int r = 0; r < 8; r++) acc[j][r] = 0.f;
    for (int ks = 0; ks < ksmax; ks++) {
      v16bf a = frag_a(&ls_p[0][0], lane & 15, Tt, ks * 32, lane);
      #pragma unroll
      for (int j = 0; j < 8; j++) {
        int nt = wave + 8 * j;
        v16bf bm = frag_b(vb, nt * 16 + (lane & 15), Tt, ks * 32, lane);
        acc[j] = wmma_bf16(a, bm, acc[j]);
      }
    }
    int rbase = (lane & 16) ? 8 : 0;
    #pragma unroll
    for (int j = 0; j < 8; j++) {
      int d = (wave + 8 * j) * 16 + (lane & 15);
      int e = d >> 6, f = d & 63, c = h * 16 + e;
      #pragma unroll
      for (int r = 0; r < 8; r++) {
        int tq = tq0 + rbase + r;
        s_out[(((long)b * Cc + c) * Tt + tq) * Ff + f] = acc[j][r];
      }
    }
  }
}

// ---------------- output projection: relu(Wo@s + bias) + LN4DCF + residual ----------------
__global__ __launch_bounds__(256) void k_oproj(
    float* __restrict__ s_io, const float* __restrict__ x,
    const __bf16* __restrict__ w, const float* __restrict__ obias,
    const float* __restrict__ g, const float* __restrict__ bb)
{
  __shared__ __bf16 ls_a[64][Cc];
  __shared__ float  ls_t[64][Cc + 1];
  __shared__ float  red[256], red2[256];
  int tid = threadIdx.x, lane = tid & 31, wave = tid >> 5;
  int t = blockIdx.x & (Tt - 1);
  int b = blockIdx.x >> 9;
  float* sb = s_io + (long)b * Cc * TF + (long)t * Ff;
  const float* xb = x + (long)b * Cc * TF + (long)t * Ff;

  for (int idx = tid; idx < 64 * Cc; idx += 256) {
    int c = idx >> 6, f = idx & 63;
    ls_a[f][c] = (__bf16)sb[(long)c * TF + f];
  }
  __syncthreads();

  {
    int mt = wave & 3, nb = wave >> 2;
    v8f acc[2];
    #pragma unroll
    for (int j = 0; j < 2; j++) {
      float bi = obias[(nb * 2 + j) * 16 + (lane & 15)];
      #pragma unroll
      for (int r = 0; r < 8; r++) acc[j][r] = bi;
    }
    #pragma unroll
    for (int ks = 0; ks < 2; ks++) {
      v16bf a = frag_a(&ls_a[0][0], mt * 16 + (lane & 15), Cc, ks * 32, lane);
      #pragma unroll
      for (int j = 0; j < 2; j++) {
        v16bf bm = frag_b(w, (nb * 2 + j) * 16 + (lane & 15), Cc, ks * 32, lane);
        acc[j] = wmma_bf16(a, bm, acc[j]);
      }
    }
    int rbase = mt * 16 + ((lane & 16) ? 8 : 0);
    #pragma unroll
    for (int j = 0; j < 2; j++) {
      int n = (nb * 2 + j) * 16 + (lane & 15);
      #pragma unroll
      for (int r = 0; r < 8; r++)
        ls_t[rbase + r][n] = fmaxf(acc[j][r], 0.0f);
    }
  }
  __syncthreads();

  float ps = 0.f, ps2 = 0.f;
  for (int j = 0; j < 16; j++) {
    int idx = tid * 16 + j;
    float v = ls_t[idx >> 6][idx & 63];
    ps += v; ps2 += v * v;
  }
  red[tid] = ps; red2[tid] = ps2;
  __syncthreads();
  for (int st = 128; st > 0; st >>= 1) {
    if (tid < st) { red[tid] += red[tid + st]; red2[tid] += red2[tid + st]; }
    __syncthreads();
  }
  float mu  = red[0] * (1.0f / 4096.0f);
  float var = red2[0] * (1.0f / 4096.0f) - mu * mu;
  float inv = fast_rcp(sqrtf(var + 1e-5f) + 1e-5f);

  for (int idx = tid; idx < 4096; idx += 256) {
    int f = idx >> 6, o = idx & 63;
    float v = (ls_t[f][o] - mu) * inv * g[o * Ff + f] + bb[o * Ff + f];
    sb[(long)o * TF + f] = v + xb[(long)o * TF + f];
  }
}

// ---------------- FreqDown conv (1,4) stride (1,2) pad (0,1) + PReLU ----------------
__global__ __launch_bounds__(256) void k_fd(
    const float* __restrict__ o, const __bf16* __restrict__ w,
    const float* __restrict__ bias, const float* __restrict__ pw,
    float* __restrict__ out)
{
  __shared__ __bf16 ls_a[32][256];   // [fo][c*4+kf]
  int tid = threadIdx.x, lane = tid & 31, wave = tid >> 5;
  int t = blockIdx.x & (Tt - 1);
  int b = blockIdx.x >> 9;
  const float* ob = o + (long)b * Cc * TF + (long)t * Ff;

  for (int idx = tid; idx < 32 * 256; idx += 256) {
    int fo = idx >> 8, kk = idx & 255, c = kk >> 2, kf = kk & 3;
    int fi = 2 * fo - 1 + kf;
    float v = (fi >= 0 && fi < Ff) ? ob[(long)c * TF + fi] : 0.0f;
    ls_a[fo][kk] = (__bf16)v;
  }
  __syncthreads();

  // GEMM M=32 N=128 K=256: wave: mt = wave&1, n-tiles (wave>>1)*2 + j
  {
    int mt = wave & 1, nb = wave >> 1;
    v8f acc[2];
    #pragma unroll
    for (int j = 0; j < 2; j++) {
      float bi = bias[(nb * 2 + j) * 16 + (lane & 15)];
      #pragma unroll
      for (int r = 0; r < 8; r++) acc[j][r] = bi;
    }
    #pragma unroll
    for (int ks = 0; ks < 8; ks++) {
      v16bf a = frag_a(&ls_a[0][0], mt * 16 + (lane & 15), 256, ks * 32, lane);
      #pragma unroll
      for (int j = 0; j < 2; j++) {
        v16bf bm = frag_b(w, (nb * 2 + j) * 16 + (lane & 15), 256, ks * 32, lane);
        acc[j] = wmma_bf16(a, bm, acc[j]);
      }
    }
    int rbase = mt * 16 + ((lane & 16) ? 8 : 0);
    #pragma unroll
    for (int j = 0; j < 2; j++) {
      int n = (nb * 2 + j) * 16 + (lane & 15);
      float pv = pw[n];
      #pragma unroll
      for (int r = 0; r < 8; r++) {
        float y = acc[j][r];
        y = (y > 0.f) ? y : pv * y;
        out[(((long)b * CO + n) * Tt + t) * FOUT + (rbase + r)] = y;
      }
    }
  }
}

// ---------------- host launcher ----------------
extern "C" void kernel_launch(void* const* d_in, const int* in_sizes, int n_in,
                              void* d_out, int out_size, void* d_ws, size_t ws_size,
                              hipStream_t stream) {
  (void)in_sizes; (void)n_in; (void)out_size; (void)ws_size;
  const float* x_in    = (const float*)d_in[0];
  const float* dw_w    = (const float*)d_in[1];
  const float* dw_b    = (const float*)d_in[2];
  const float* ln_w    = (const float*)d_in[3];
  const float* ln_b    = (const float*)d_in[4];
  const float* pw1_w   = (const float*)d_in[5];
  const float* pw1_b   = (const float*)d_in[6];
  const float* pw2_w   = (const float*)d_in[7];
  const float* pw2_b   = (const float*)d_in[8];
  const float* q_w     = (const float*)d_in[9];
  const float* q_g     = (const float*)d_in[10];
  const float* q_b     = (const float*)d_in[11];
  const float* k_w     = (const float*)d_in[12];
  const float* k_g     = (const float*)d_in[13];
  const float* k_b     = (const float*)d_in[14];
  const float* v_w     = (const float*)d_in[15];
  const float* v_g     = (const float*)d_in[16];
  const float* v_b     = (const float*)d_in[17];
  const float* o_w     = (const float*)d_in[18];
  const float* o_bias  = (const float*)d_in[19];
  const float* o_g     = (const float*)d_in[20];
  const float* o_b     = (const float*)d_in[21];
  const float* fd_w    = (const float*)d_in[22];
  const float* fd_b    = (const float*)d_in[23];
  const float* prelu_w = (const float*)d_in[24];

  char* ws = (char*)d_ws;
  const size_t NX = (size_t)Bq * Cc * Tt * Ff;       // 16,777,216
  float*  buf0 = (float*)(ws);                       // 64 MB: working x
  float*  buf1 = (float*)(ws + NX * 4);              // 64 MB: dw out / attn s
  __bf16* qbuf = (__bf16*)(ws + NX * 8);             // 32 MB
  __bf16* kbuf = (__bf16*)(ws + NX * 8 + NX * 2);    // 32 MB
  __bf16* vtbf = (__bf16*)(ws + NX * 8 + NX * 4);    // 32 MB (transposed V)
  __bf16* w1bf = (__bf16*)(ws + NX * 8 + NX * 6);    // weights region
  __bf16* w2bf = w1bf + NLAY * RC * Cc;
  __bf16* qwbf = w2bf + NLAY * Cc * RC;
  __bf16* kwbf = qwbf + Cc * Cc;
  __bf16* vwbf = kwbf + Cc * Cc;
  __bf16* owbf = vwbf + Cc * Cc;
  __bf16* fdwb = owbf + Cc * Cc;

  hipMemcpyAsync(buf0, x_in, NX * sizeof(float), hipMemcpyDeviceToDevice, stream);

  auto cvt = [&](const float* s, __bf16* d, int n) {
    k_cvt<<<(n + 255) / 256, 256, 0, stream>>>(s, d, n);
  };
  cvt(pw1_w, w1bf, NLAY * RC * Cc);
  cvt(pw2_w, w2bf, NLAY * Cc * RC);
  cvt(q_w, qwbf, Cc * Cc);
  cvt(k_w, kwbf, Cc * Cc);
  cvt(v_w, vwbf, Cc * Cc);
  cvt(o_w, owbf, Cc * Cc);
  cvt(fd_w, fdwb, CO * Cc * 4);

  // ---- TFCM: 6 residual blocks ----
  for (int i = 0; i < NLAY; i++) {
    int dil = 1 << i;
    k_tfcm_dw<<<Bq * Cc * Tt, 64, 0, stream>>>(
        buf0, buf1, dw_w + (size_t)i * Cc * 14, dw_b + (size_t)i * Cc, dil);
    k_tfcm_mlp<<<4096, 256, 0, stream>>>(
        buf1, buf0, ln_w + (size_t)i * Cc, ln_b + (size_t)i * Cc,
        w1bf + (size_t)i * RC * Cc, pw1_b + (size_t)i * RC,
        w2bf + (size_t)i * Cc * RC, pw2_b + (size_t)i * Cc);
  }

  // ---- projections ----
  k_proj<<<Bq * Tt, 256, 0, stream>>>(buf0, qwbf, q_g, q_b, qbuf, 0);
  k_proj<<<Bq * Tt, 256, 0, stream>>>(buf0, kwbf, k_g, k_b, kbuf, 0);
  k_proj<<<Bq * Tt, 256, 0, stream>>>(buf0, vwbf, v_g, v_b, vtbf, 1);

  // ---- attention ----
  k_attn<<<Bq * Hh * (Tt / 16), 256, 0, stream>>>(qbuf, kbuf, vtbf, buf1);

  // ---- output projection + LN + residual (in place on buf1) ----
  k_oproj<<<Bq * Tt, 256, 0, stream>>>(buf1, buf0, owbf, o_bias, o_g, o_b);

  // ---- freq down + PReLU ----
  k_fd<<<Bq * Tt, 256, 0, stream>>>(buf1, fdwb, fd_b, prelu_w, (float*)d_out);
}
